// CSOCSSC_V15_5_61890478735633
// MI455X (gfx1250) — matrix operand order repaired and checked by reading, not compiled, not run on hardware
//
#include <hip/hip_runtime.h>
#include <stdint.h>

typedef __attribute__((ext_vector_type(16))) __bf16 v16bf;
typedef __attribute__((ext_vector_type(8)))  float  v8f;

#define NRES   8192
#define DM     256
#define EPS_D  1e-6f
#define EPS_N  1e-8f

// padded K-tile row stride (dwords): 16B-aligned, all 16 lane offsets distinct mod 64 banks
#define KSTRIDE 36

// base-2 folded constants
#define C_K12   0.12022458674074695f   // 1/(12*ln2)
#define C_KAPPA 0.14426950408889634f   // 0.1/ln2
#define C_HB    4.003741362621499f     // sqrt(1/ln2)/0.3
#define C_BUR   0.07213475204444817f   // (1/20)/ln2

__device__ __forceinline__ float fsqrt(float x)  { return __builtin_amdgcn_sqrtf(x); }
__device__ __forceinline__ float flog2(float x)  { return __builtin_amdgcn_logf(x); }
__device__ __forceinline__ float fexp2(float x)  { return __builtin_amdgcn_exp2f(x); }
__device__ __forceinline__ float frcp(float x)   { return __builtin_amdgcn_rcpf(x); }

__device__ __forceinline__ uint32_t pack_bf16(float a, float b) {
  // RNE round both to bf16, pack {b,a} with one v_perm_b32
  uint32_t ua = __builtin_bit_cast(uint32_t, a);
  uint32_t ub = __builtin_bit_cast(uint32_t, b);
  ua += 0x7FFFu + ((ua >> 16) & 1u);
  ub += 0x7FFFu + ((ub >> 16) & 1u);
  return __builtin_amdgcn_perm(ub, ua, 0x07060302u); // {ub[31:16], ua[31:16]}
}

// ---------------------------------------------------------------------------
// Prep: transpose latent -> bf16 latT[c][r], reconstruct backbone N/O atoms.
// ---------------------------------------------------------------------------
__global__ __launch_bounds__(256) void prep_kernel(
    const float* __restrict__ coords, const float* __restrict__ latent,
    uint32_t* __restrict__ latT32, float* __restrict__ Nat, float* __restrict__ Oat) {
  long T = (long)blockIdx.x * 256 + threadIdx.x;
  if (T < (long)NRES * DM / 2) {
    int c  = (int)(T / (NRES / 2));
    int r2 = (int)(T % (NRES / 2));
    int r  = r2 * 2;
    float a = latent[(long)r * DM + c];
    float b = latent[(long)(r + 1) * DM + c];
    latT32[(long)c * (NRES / 2) + r2] = pack_bf16(a, b);
  }
  if (T < NRES) {
    int i = (int)T;
    float cx = coords[i*3+0], cy = coords[i*3+1], cz = coords[i*3+2];
    float nx = 0.f, ny = 0.f, nz = 0.f;
    if (i > 0) {
      float vx = cx - coords[(i-1)*3+0];
      float vy = cy - coords[(i-1)*3+1];
      float vz = cz - coords[(i-1)*3+2];
      float inr = frcp(fmaxf(fsqrt(vx*vx+vy*vy+vz*vz), 1e-12f));
      nx = cx - 1.45f*vx*inr; ny = cy - 1.45f*vy*inr; nz = cz - 1.45f*vz*inr;
    }
    Nat[i*3+0]=nx; Nat[i*3+1]=ny; Nat[i*3+2]=nz;
    float ox, oy, oz;
    if (i < NRES-1) {
      float vx = coords[(i+1)*3+0]-cx, vy = coords[(i+1)*3+1]-cy, vz = coords[(i+1)*3+2]-cz;
      float inr = frcp(fmaxf(fsqrt(vx*vx+vy*vy+vz*vz), 1e-12f));
      ox = cx + 1.52f*vx*inr; oy = cy + 1.52f*vy*inr; oz = cz + 1.52f*vz*inr;
    } else {
      float px = coords[(NRES-2)*3+0], py = coords[(NRES-2)*3+1], pz = coords[(NRES-2)*3+2];
      float vx = cx-px, vy = cy-py, vz = cz-pz;
      float inr = frcp(fmaxf(fsqrt(vx*vx+vy*vy+vz*vz), 1e-12f));
      ox = px + 1.52f*vx*inr; oy = py + 1.52f*vy*inr; oz = pz + 1.52f*vz*inr;
    }
    Oat[i*3+0]=ox; Oat[i*3+1]=oy+1.24f; Oat[i*3+2]=oz;
  }
}

// ---------------------------------------------------------------------------
// Fused contact-kernel construction + K @ latent via WMMA.
// One workgroup = 16 output rows x all 256 cols. 8 waves, 2 16x16 tiles each.
// K chunk = 64: each thread produces 4 K entries, each wave issues 4 WMMAs
// per barrier. Double-buffered, bank-conflict-free padded LDS tile.
// ---------------------------------------------------------------------------
__global__ __launch_bounds__(256) void diffuse_kernel(
    const float* __restrict__ coords, const float* __restrict__ alpha,
    const uint32_t* __restrict__ latT32, float* __restrict__ out) {
  __shared__ uint32_t ktile[2][16 * KSTRIDE];  // pair p of row m at [m*KSTRIDE + p]
  __shared__ float red[256];
  __shared__ float invRow[16];

  const int tid = threadIdx.x;
  const int rowBase = blockIdx.x * 16;

  // ---- producer role: thread computes 4 K entries per chunk (pairs pk, pk+16)
  const int pm = tid >> 4;            // tile row 0..15
  const int pk = tid & 15;            // pair 0..15 (and pk+16)
  const int gr = rowBase + pm;
  const float rx = coords[gr*3+0], ry = coords[gr*3+1], rz = coords[gr*3+2];
  const float ar = alpha[gr];
  float rowpart = 0.f;

  // ---- consumer role: wave does 2 WMMA column tiles x 2 sub-chunks
  const int wave = tid >> 5;          // 0..7
  const int lane = tid & 31;
  const int half = lane >> 4;
  const int mi = lane & 15;
  const int c0 = wave * 32;
  const int c1 = c0 + 16;

  v8f acc0 = {}; v8f acc1 = {};

  for (int it = 0; it < NRES / 64; ++it) {
    const int kc  = it * 64;
    const int buf = it & 1;
    // produce 4 K entries (1 v_log + 1 v_exp each, branchless diag mask)
    {
      int g0 = kc + pk * 2;        // pair pk   -> k = 2pk, 2pk+1
      int g2 = g0 + 32;            // pair pk+16 -> k = 2pk+32, 2pk+33
      float kv[4];
#pragma unroll
      for (int e = 0; e < 4; ++e) {
        int g = (e < 2) ? (g0 + e) : (g2 + (e - 2));
        float dx = rx - coords[g*3+0], dy = ry - coords[g*3+1], dz = rz - coords[g*3+2];
        float d = fsqrt(dx*dx + dy*dy + dz*dz);
        float a = 0.5f*(ar + alpha[g]);
        float m = (g != gr) ? 1.f : 0.f;
        kv[e] = m * fexp2(-a * flog2(d + EPS_D) - d * C_K12);
      }
      rowpart += (kv[0] + kv[1]) + (kv[2] + kv[3]);
      ktile[buf][pm*KSTRIDE + pk]      = pack_bf16(kv[0], kv[1]);
      ktile[buf][pm*KSTRIDE + 16 + pk] = pack_bf16(kv[2], kv[3]);
    }
    __syncthreads();

    // consume: 2 sub-chunks of 32, 2 column tiles each
#pragma unroll
    for (int s = 0; s < 2; ++s) {
      union { uint32_t u[8]; v16bf v; } A, B0, B1;
#pragma unroll
      for (int v = 0; v < 8; ++v) {
        // A (16x32 bf16): lanes 0-15 hold K {0..7,16..23}, lanes 16-31 {8..15,24..31}
        int ka = ((v & 3) * 2) + ((v & 4) ? 16 : 0) + half * 8;
        A.u[v] = ktile[buf][mi*KSTRIDE + s*16 + (ka >> 1)];
        // B (32x16 bf16): lanes 0-15 hold K {0..7,16..23}, lanes 16-31 {8..15,24..31}
        int gk = kc + s*32 + ((v & 3) * 2) + ((v & 4) ? 16 : 0) + half * 8;
        B0.u[v] = latT32[(size_t)(c0+mi)*4096 + (gk >> 1)];
        B1.u[v] = latT32[(size_t)(c1+mi)*4096 + (gk >> 1)];
      }
      acc0 = __builtin_amdgcn_wmma_f32_16x16x32_bf16(false, A.v, false, B0.v, (short)0, acc0, false, false);
      acc1 = __builtin_amdgcn_wmma_f32_16x16x32_bf16(false, A.v, false, B1.v, (short)0, acc1, false, false);
    }
    // no trailing barrier: next iteration writes the other LDS buffer, and this
    // iteration's ds_loads are consumed before this wave reaches the next barrier
  }

  // row-sum reduce (fixed order), normalize, store
  red[tid] = rowpart;
  __syncthreads();
  if (tid < 16) {
    float s = 0.f;
#pragma unroll
    for (int j = 0; j < 16; ++j) s += red[tid*16 + j];
    invRow[tid] = frcp(s + EPS_N);
  }
  __syncthreads();
#pragma unroll
  for (int r = 0; r < 8; ++r) {
    int M = half*8 + r;
    float sc = invRow[M];
    size_t base = (size_t)(rowBase + M) * DM;
    out[base + c0 + mi] = acc0[r] * sc;
    out[base + c1 + mi] = acc1[r] * sc;
  }
}

// ---------------------------------------------------------------------------
// Pairwise energy terms: 1 row/thread, j-range split 4 ways over blockIdx.y.
// ---------------------------------------------------------------------------
__global__ __launch_bounds__(256) void energy_kernel(
    const float* __restrict__ coords, const float* __restrict__ q,
    const float* __restrict__ Nat, const float* __restrict__ Oat,
    float* __restrict__ partial_el, float* __restrict__ partial_hb,
    float* __restrict__ dens) {
  const int tid = threadIdx.x;
  const int i = blockIdx.x * 256 + tid;
  const int j0 = blockIdx.y * (NRES/4);
  const int j1 = j0 + (NRES/4);
  const float cx = coords[i*3], cy = coords[i*3+1], cz = coords[i*3+2];
  const float ox = Oat[i*3],   oy = Oat[i*3+1],   oz = Oat[i*3+2];
  const float qi80 = q[i] * 0.0125f;   // qi / 80
  float ael = 0.f, ahb = 0.f, density = 0.f;
  for (int j = j0; j < j1; ++j) {
    float dx = cx - coords[j*3], dy = cy - coords[j*3+1], dz = cz - coords[j*3+2];
    float d = fsqrt(dx*dx + dy*dy + dz*dz);
    ael += qi80 * q[j] * fexp2(-d * C_KAPPA) * frcp(d + EPS_D);
    density += (d < 10.f) ? 1.f : 0.f;
    float nx = ox - Nat[j*3], ny = oy - Nat[j*3+1], nz = oz - Nat[j*3+2];
    float dh = fsqrt(nx*nx + ny*ny + nz*nz);
    float u = (dh - 2.95f) * C_HB;
    float w = (dh > 2.5f && dh < 3.5f) ? 1.f : 0.f;
    ahb -= w * fexp2(-u * u);
  }
  dens[(size_t)blockIdx.y * NRES + i] = density;

  __shared__ float red[256];
  red[tid] = ael; __syncthreads();
  for (int s = 128; s > 0; s >>= 1) { if (tid < s) red[tid] += red[tid+s]; __syncthreads(); }
  const int bidx = blockIdx.y * gridDim.x + blockIdx.x;
  if (tid == 0) partial_el[bidx] = red[0];
  __syncthreads();
  red[tid] = ahb; __syncthreads();
  for (int s = 128; s > 0; s >>= 1) { if (tid < s) red[tid] += red[tid+s]; __syncthreads(); }
  if (tid == 0) partial_hb[bidx] = red[0];
}

// ---------------------------------------------------------------------------
// Final combine: sum partials (fixed order), solvent term per row, one scalar.
// ---------------------------------------------------------------------------
__global__ __launch_bounds__(256) void finalize_kernel(
    const float* __restrict__ hydro,
    const float* __restrict__ partial_el, const float* __restrict__ partial_hb,
    const float* __restrict__ dens, float* __restrict__ out_scalar) {
  const int tid = threadIdx.x;
  __shared__ float red[256];
  __shared__ float sums[2];

  red[tid] = (tid < 128) ? partial_el[tid] : 0.f;
  __syncthreads();
  for (int s = 128; s > 0; s >>= 1) { if (tid < s) red[tid] += red[tid+s]; __syncthreads(); }
  if (tid == 0) sums[0] = red[0];
  __syncthreads();

  red[tid] = (tid < 128) ? partial_hb[tid] : 0.f;
  __syncthreads();
  for (int s = 128; s > 0; s >>= 1) { if (tid < s) red[tid] += red[tid+s]; __syncthreads(); }
  if (tid == 0) sums[1] = red[0];
  __syncthreads();

  float sv = 0.f;
  for (int i = tid; i < NRES; i += 256) {
    float dsum = dens[i] + dens[NRES + i] + dens[2*NRES + i] + dens[3*NRES + i];
    float burial = 1.f - fexp2(-dsum * C_BUR);
    float h = hydro[i];
    sv += (h > 0.f) ? h * burial : h * (1.f - burial);
  }
  red[tid] = sv; __syncthreads();
  for (int s = 128; s > 0; s >>= 1) { if (tid < s) red[tid] += red[tid+s]; __syncthreads(); }
  if (tid == 0) {
    const float invNN = 1.f / ((float)NRES * (float)NRES);
    out_scalar[0] = sums[0]*invNN + sums[1]*invNN + red[0]*(1.f/(float)NRES);
  }
}

// ---------------------------------------------------------------------------
extern "C" void kernel_launch(void* const* d_in, const int* in_sizes, int n_in,
                              void* d_out, int out_size, void* d_ws, size_t ws_size,
                              hipStream_t stream) {
  const float* coords = (const float*)d_in[0];   // [8192,3]
  const float* latent = (const float*)d_in[1];   // [8192,256]
  const float* alpha  = (const float*)d_in[2];   // [8192]
  const float* q      = (const float*)d_in[3];   // [8192]
  const float* hydro  = (const float*)d_in[4];   // [8192]
  float* out = (float*)d_out;                    // [8192*256] + 1 scalar

  char* ws = (char*)d_ws;
  uint32_t* latT32  = (uint32_t*)(ws);                              // 4 MB bf16 [256][8192]
  float* Nat        = (float*)(ws + 4194304);                       // 96 KB
  float* Oat        = (float*)(ws + 4194304 + 98304);               // 96 KB
  float* partial_el = (float*)(ws + 4194304 + 196608);              // 512 B
  float* partial_hb = (float*)(ws + 4194304 + 196608 + 512);        // 512 B
  float* dens       = (float*)(ws + 4194304 + 196608 + 1024);       // 128 KB

  prep_kernel<<<(NRES*DM)/256, 256, 0, stream>>>(coords, latent, latT32, Nat, Oat);
  diffuse_kernel<<<NRES/16, 256, 0, stream>>>(coords, alpha, latT32, out);
  energy_kernel<<<dim3(NRES/256, 4), 256, 0, stream>>>(coords, q, Nat, Oat,
                                                       partial_el, partial_hb, dens);
  finalize_kernel<<<1, 256, 0, stream>>>(hydro, partial_el, partial_hb, dens,
                                         out + (size_t)NRES * DM);
}
